// BrainGCN_8289286882026
// MI455X (gfx1250) — matrix operand order repaired and checked by reading, not compiled, not run on hardware
//
#include <hip/hip_runtime.h>
#include <hip/hip_bf16.h>

// ---------------------------------------------------------------------------
// BrainGCN on gfx1250: bf16 WMMA GEMMs (packed B fragments, b128 loads)
//                     + fp32 L2-resident atomic scatter for edge aggregation.
// ---------------------------------------------------------------------------

typedef __attribute__((ext_vector_type(16))) __bf16 bf16x16;
typedef __attribute__((ext_vector_type(8)))  float  f32x8;

static inline int cdiv(int a, int b) { return (a + b - 1) / b; }

// --- degree / normalization -------------------------------------------------

__global__ void k_fill1(float* __restrict__ p, int n) {
    int i = blockIdx.x * blockDim.x + threadIdx.x;
    if (i < n) p[i] = 1.0f;  // self-loop contributes 1 to degree
}

__global__ void k_deg(const long long* __restrict__ dst, float* __restrict__ deg, int E) {
    int i = blockIdx.x * blockDim.x + threadIdx.x;
    if (i < E) atomicAdd(&deg[(size_t)dst[i]], 1.0f);
}

__global__ void k_rsqrt_inplace(float* __restrict__ p, int n) {
    int i = blockIdx.x * blockDim.x + threadIdx.x;
    if (i < n) p[i] = rsqrtf(p[i]);
}

// --- fp32 -> bf16 cast (activations) ---------------------------------------

__global__ void k_cast_bf16(const float* __restrict__ in, __bf16* __restrict__ out, int n) {
    int i = blockIdx.x * blockDim.x + threadIdx.x;
    if (i < n) out[i] = (__bf16)in[i];
}

// --- pack W[K x F] (fp32 row-major) into CDNA5 WMMA B-fragment layout -------
// Bp element index = ((nt*NK + ks)*32 + lane)*16 + e
// holds W[ks*32 + 8*(lane>>4) + (e&7) + 16*(e>>3)][nt*16 + (lane&15)] as bf16.
// Result: each lane's 16 B elements are 32 contiguous bytes (2 x b128 loads).

__global__ void k_pack_w(const float* __restrict__ W, __bf16* __restrict__ Bp,
                         int K, int F) {
    int tid = blockIdx.x * blockDim.x + threadIdx.x;
    if (tid >= K * F) return;
    int e    = tid & 15;
    int lane = (tid >> 4) & 31;
    int rest = tid >> 9;               // = nt*NK + ks
    int NK   = K >> 5;
    int ks   = rest % NK;
    int nt   = rest / NK;
    int kk   = ks * 32 + 8 * (lane >> 4) + (e & 7) + ((e >> 3) << 4);
    int col  = nt * 16 + (lane & 15);
    Bp[tid] = (__bf16)W[(size_t)kk * F + col];
}

// --- WMMA GEMM: C[M x F] = A[M x K] @ Bp(packed), fp32 accumulate -----------
// One wave computes a 16x32 output block (two n-tiles) so the A fragment is
// reused across two v_wmma ops. K-loop fully unrolled (K, F compile-time).
//
// CDNA5 16-bit A-matrix 16x32 lane layout (ISA 7.12.2):
//   lane = 16*kgrp + m ; element e holds K = 8*kgrp + (e>=8?16:0) + (e&7)
//   -> two contiguous 8-elem runs per lane: [8k, 8k+8) and [8k+16, 8k+24)
// C/D 16x16 f32: vgpr r, lane l -> M = 8*(l>>4) + r, N = l&15

template <int K, int F>
__global__ void k_gemm_bf16_wmma(const __bf16* __restrict__ A,
                                 const __bf16* __restrict__ Bp,
                                 float* __restrict__ C, int mTiles) {
    constexpr int NK  = K / 32;        // k-steps
    constexpr int NP  = F / 32;        // n-tile pairs per row of tiles
    int wave = (int)((blockIdx.x * blockDim.x + threadIdx.x) >> 5);
    int lane = threadIdx.x & 31;
    if (wave >= mTiles * NP) return;   // wave-uniform: EXEC stays all-ones

    int mt   = wave / NP;
    int np   = wave - mt * NP;
    int ml   = lane & 15;
    int kgrp = lane >> 4;

    const uint4* Arow = (const uint4*)(A + (size_t)(mt * 16 + ml) * K + 8 * kgrp);
    const uint4* Bq   = (const uint4*)Bp;
    const size_t b0   = ((size_t)(2 * np)     * NK * 32 + lane) * 2;
    const size_t b1   = ((size_t)(2 * np + 1) * NK * 32 + lane) * 2;

    f32x8 acc0 = {}, acc1 = {};
#pragma unroll
    for (int ks = 0; ks < NK; ++ks) {
        union { uint4 q[2]; bf16x16 v; } a, w0, w1;
        a.q[0]  = Arow[ks * 4 + 0];    // K = k0+8k .. +7   (16B aligned)
        a.q[1]  = Arow[ks * 4 + 2];    // K = k0+8k+16 .. +23
        w0.q[0] = Bq[b0 + ks * 64 + 0];
        w0.q[1] = Bq[b0 + ks * 64 + 1];
        w1.q[0] = Bq[b1 + ks * 64 + 0];
        w1.q[1] = Bq[b1 + ks * 64 + 1];
        acc0 = __builtin_amdgcn_wmma_f32_16x16x32_bf16(
            false, a.v, false, w0.v, (short)0, acc0, false, false);
        acc1 = __builtin_amdgcn_wmma_f32_16x16x32_bf16(
            false, a.v, false, w1.v, (short)0, acc1, false, false);
    }

    int orow = mt * 16 + 8 * kgrp;
    int col0 = np * 32 + ml;
#pragma unroll
    for (int r = 0; r < 8; ++r) {
        C[(size_t)(orow + r) * F + col0]      = acc0[r];
        C[(size_t)(orow + r) * F + col0 + 16] = acc1[r];
    }
}

// --- GCN aggregation --------------------------------------------------------

// agg[i,f] = xw[i,f] * dinv[i]^2 + bias[f]   (self-loop term + bias folded in)
__global__ void k_agg_init(const float* __restrict__ xw, const float* __restrict__ dinv,
                           const float* __restrict__ bias, float* __restrict__ agg,
                           int N, int F) {
    int tid = blockIdx.x * blockDim.x + threadIdx.x;
    if (tid >= N * F) return;
    int i = tid / F;
    int f = tid - i * F;
    float di = dinv[i];
    agg[tid] = xw[tid] * di * di + bias[f];
}

// per edge: agg[dst,:] += xw[src,:] * dinv[src]*dinv[dst]
// one thread handles 4 features; segShift = log2(F/4) (5 for F=128, 4 for F=64)
__global__ void k_scatter(const long long* __restrict__ src, const long long* __restrict__ dst,
                          const float* __restrict__ xw, const float* __restrict__ dinv,
                          float* __restrict__ agg, int E, int segShift, int F) {
    int tid = blockIdx.x * blockDim.x + threadIdx.x;
    int e   = tid >> segShift;
    if (e >= E) return;
    int sg  = tid & ((1 << segShift) - 1);

    long long s = src[e];
    long long d = dst[e];
    float coef = dinv[(size_t)s] * dinv[(size_t)d];

    const float4 v = ((const float4*)(xw + (size_t)s * F))[sg];
    float* o = agg + (size_t)d * F + sg * 4;
    atomicAdd(o + 0, v.x * coef);
    atomicAdd(o + 1, v.y * coef);
    atomicAdd(o + 2, v.z * coef);
    atomicAdd(o + 3, v.w * coef);
}

// --- activations / epilogues ------------------------------------------------

__global__ void k_tanh_to_bf16(const float* __restrict__ in, __bf16* __restrict__ out, int n) {
    int i = blockIdx.x * blockDim.x + threadIdx.x;
    if (i < n) out[i] = (__bf16)tanhf(in[i]);
}

// h3[i,f] = tanh(in[i,f] + bias[f])  (kept fp32 for the final dot)
__global__ void k_bias_tanh(const float* __restrict__ in, const float* __restrict__ bias,
                            float* __restrict__ out, int N, int F) {
    int tid = blockIdx.x * blockDim.x + threadIdx.x;
    if (tid >= N * F) return;
    int f = tid & (F - 1);
    out[tid] = tanhf(in[tid] + bias[f]);
}

// out[i] = h3[i,:] . W4[:,0] + b4
__global__ void k_fc4(const float* __restrict__ h3, const float* __restrict__ W4,
                      const float* __restrict__ b4, float* __restrict__ out, int N) {
    int i = blockIdx.x * blockDim.x + threadIdx.x;
    if (i >= N) return;
    float sum = b4[0];
    const float* row = h3 + (size_t)i * 64;
#pragma unroll
    for (int f = 0; f < 64; ++f) sum = fmaf(row[f], W4[f], sum);
    out[i] = sum;
}

// ---------------------------------------------------------------------------

extern "C" void kernel_launch(void* const* d_in, const int* in_sizes, int n_in,
                              void* d_out, int out_size, void* d_ws, size_t ws_size,
                              hipStream_t stream) {
    const float*     x   = (const float*)d_in[0];
    const long long* ei  = (const long long*)d_in[1];   // [2, E] int64
    const float*     W1  = (const float*)d_in[2];
    const float*     b1  = (const float*)d_in[3];
    const float*     W2  = (const float*)d_in[4];
    const float*     b2  = (const float*)d_in[5];
    const float*     W3  = (const float*)d_in[6];
    const float*     b3  = (const float*)d_in[7];
    const float*     W4  = (const float*)d_in[8];
    const float*     b4  = (const float*)d_in[9];
    float* out = (float*)d_out;

    const int F_IN = 128, H1 = 128, H2 = 64, H3 = 64;
    const int N = in_sizes[0] / F_IN;        // 50000
    const int E = in_sizes[1] / 2;           // 800000
    const long long* src = ei;
    const long long* dst = ei + E;

    // workspace bump allocator (256B aligned)
    char* p = (char*)d_ws;
    auto alloc = [&](size_t bytes) -> void* {
        void* r = (void*)p;
        p += (bytes + 255) & ~(size_t)255;
        return r;
    };
    float*  dinv = (float*)alloc((size_t)N * sizeof(float));
    __bf16* xbf  = (__bf16*)alloc((size_t)N * F_IN * sizeof(__bf16));
    float*  xw   = (float*)alloc((size_t)N * F_IN * sizeof(float));   // reused F<=128
    float*  agg  = (float*)alloc((size_t)N * F_IN * sizeof(float));   // reused F<=128
    __bf16* hbf  = (__bf16*)alloc((size_t)N * F_IN * sizeof(__bf16)); // reused
    float*  h3   = (float*)alloc((size_t)N * H3 * sizeof(float));
    __bf16* w1p  = (__bf16*)alloc((size_t)F_IN * H1 * sizeof(__bf16));
    __bf16* w2p  = (__bf16*)alloc((size_t)H1 * H2 * sizeof(__bf16));
    __bf16* w3p  = (__bf16*)alloc((size_t)H2 * H3 * sizeof(__bf16));

    const int T = 256;

    // 1) symmetric normalization coefficients
    k_fill1<<<cdiv(N, T), T, 0, stream>>>(dinv, N);
    k_deg<<<cdiv(E, T), T, 0, stream>>>(dst, dinv, E);
    k_rsqrt_inplace<<<cdiv(N, T), T, 0, stream>>>(dinv, N);

    // 2) bf16 activation cast + weight fragment packing
    k_cast_bf16<<<cdiv(N * F_IN, T), T, 0, stream>>>(x, xbf, N * F_IN);
    k_pack_w<<<cdiv(F_IN * H1, T), T, 0, stream>>>(W1, w1p, F_IN, H1);
    k_pack_w<<<cdiv(H1 * H2, T), T, 0, stream>>>(W2, w2p, H1, H2);
    k_pack_w<<<cdiv(H2 * H3, T), T, 0, stream>>>(W3, w3p, H2, H3);

    const int mT = N / 16;                   // 3125, exact (no tail tiles)

    // 3) layer 1: xw = x @ W1 ; agg = scatter + self ; h1 = tanh -> bf16
    {
        int waves = mT * (H1 / 32);
        k_gemm_bf16_wmma<F_IN, H1><<<cdiv(waves * 32, T), T, 0, stream>>>(xbf, w1p, xw, mT);
        k_agg_init<<<cdiv(N * H1, T), T, 0, stream>>>(xw, dinv, b1, agg, N, H1);
        k_scatter<<<cdiv(E * (H1 / 4), T), T, 0, stream>>>(src, dst, xw, dinv, agg, E, 5, H1);
        k_tanh_to_bf16<<<cdiv(N * H1, T), T, 0, stream>>>(agg, hbf, N * H1);
    }
    // 4) layer 2: xw2 = h1 @ W2 ; aggregate ; h2 = tanh -> bf16
    {
        int waves = mT * (H2 / 32);
        k_gemm_bf16_wmma<H1, H2><<<cdiv(waves * 32, T), T, 0, stream>>>(hbf, w2p, xw, mT);
        k_agg_init<<<cdiv(N * H2, T), T, 0, stream>>>(xw, dinv, b2, agg, N, H2);
        k_scatter<<<cdiv(E * (H2 / 4), T), T, 0, stream>>>(src, dst, xw, dinv, agg, E, 4, H2);
        k_tanh_to_bf16<<<cdiv(N * H2, T), T, 0, stream>>>(agg, hbf, N * H2);
    }
    // 5) fc3: h3 = tanh(h2 @ W3 + b3) (fp32 result)
    {
        int waves = mT * (H3 / 32);
        k_gemm_bf16_wmma<H2, H3><<<cdiv(waves * 32, T), T, 0, stream>>>(hbf, w3p, xw, mT);
        k_bias_tanh<<<cdiv(N * H3, T), T, 0, stream>>>(xw, b3, h3, N, H3);
    }
    // 6) fc4: out = h3 @ W4 + b4
    k_fc4<<<cdiv(N, T), T, 0, stream>>>(h3, W4, b4, out, N);
}
